// SelfAttention_27195732918936
// MI455X (gfx1250) — matrix operand order repaired and checked by reading, not compiled, not run on hardware
//
#include <hip/hip_runtime.h>

// ---------------------------------------------------------------------------
// MI455X (gfx1250) self-attention prefill.
// Compute-bound (~412 GFLOP vs ~13us of HBM traffic at 23.3 TB/s), so every
// matmul goes through V_WMMA_F32_16X16X32_BF16 (wave32, f32 accumulate).
// v3: 2-deep global->reg->LDS pipelines (1 sync/iter, double-buffered LDS)
//     in both GEMM and attention; softmax row reductions via DPP16 ROW_XMASK
//     (pure VALU) instead of ds_bpermute; K/V fragments preloaded in arrays
//     so ds_load clauses drain ahead of the WMMA chains.
// ---------------------------------------------------------------------------

typedef __attribute__((ext_vector_type(16))) __bf16 v16bf;
typedef __attribute__((ext_vector_type(8)))  __bf16 v8bf;
typedef __attribute__((ext_vector_type(4)))  __bf16 v4bf;
typedef __attribute__((ext_vector_type(8)))  float  v8f;

constexpr int BATCH = 4;
constexpr int SEQ   = 1024;
constexpr int DIM   = 4096;
constexpr int HQ    = 32;
constexpr int HKV   = 8;
constexpr int HD    = 128;

#define DEV __device__ __forceinline__

DEV __bf16 f2bf(float f) {           // round-to-nearest-even f32 -> bf16
  unsigned u = __float_as_uint(f);
  unsigned r = (u + 0x7fffu + ((u >> 16) & 1u)) >> 16;
  unsigned short s = (unsigned short)r;
  __bf16 b; __builtin_memcpy(&b, &s, 2); return b;
}
DEV float bf2f(__bf16 b) {
  unsigned short s; __builtin_memcpy(&s, &b, 2);
  return __uint_as_float(((unsigned)s) << 16);
}

union V16U { v16bf v; v8bf h[2]; };

// A-matrix 16x32 bf16 fragment (ISA 7.12.2): lane holds row M=lane%16;
// halves [k0..k0+7] in v[0..7], [k0+16..k0+23] in v[8..15], k0=(lane<16?0:8).
DEV v16bf frag_a(const __bf16* row, int lane) {
  int k0 = (lane < 16) ? 0 : 8;
  V16U u;
  u.h[0] = *(const v8bf*)(row + k0);
  u.h[1] = *(const v8bf*)(row + k0 + 16);
  return u.v;
}
// B-matrix 32x16 bf16 fragment: lane holds col N=lane%16; same K order as A:
// [k0..k0+7] in v[0..7], [k0+16..k0+23] in v[8..15], k0=(lane<16?0:8).
DEV v16bf frag_b(const __bf16* row, int lane) {
  int k0 = (lane < 16) ? 0 : 8;
  V16U u;
  u.h[0] = *(const v8bf*)(row + k0);
  u.h[1] = *(const v8bf*)(row + k0 + 16);
  return u.v;
}
DEV v8f wmma_bf16(v16bf a, v16bf b, v8f c) {
  return __builtin_amdgcn_wmma_f32_16x16x32_bf16(false, a, false, b,
                                                 (short)0, c, false, false);
}

// ---- DPP16 ROW_XMASK lane-xor reductions over each 16-lane half (VALU only)
template <int CTRL>
DEV float dppmov(float x) {
  return __int_as_float(__builtin_amdgcn_update_dpp(
      0, __float_as_int(x), CTRL, 0xf, 0xf, true));
}
DEV float rmax16(float t) {          // all 16 lanes of each half get the max
  t = fmaxf(t, dppmov<0x161>(t));    // row_xmask:1
  t = fmaxf(t, dppmov<0x162>(t));    // row_xmask:2
  t = fmaxf(t, dppmov<0x164>(t));    // row_xmask:4
  t = fmaxf(t, dppmov<0x168>(t));    // row_xmask:8
  return t;
}
DEV float rsum16(float t) {
  t += dppmov<0x161>(t);
  t += dppmov<0x162>(t);
  t += dppmov<0x164>(t);
  t += dppmov<0x168>(t);
  return t;
}

// ---------------------------------------------------------------------------
// f32 -> bf16 cast (vectorized)
// ---------------------------------------------------------------------------
__global__ void cast_bf16_kernel(const float* __restrict__ in,
                                 __bf16* __restrict__ out, int n4) {
  int i = blockIdx.x * 256 + threadIdx.x;
  if (i >= n4) return;
  float4 f = ((const float4*)in)[i];
  v4bf o;
  o[0] = f2bf(f.x); o[1] = f2bf(f.y); o[2] = f2bf(f.z); o[3] = f2bf(f.w);
  *(v4bf*)(out + (size_t)i * 4) = o;
}

// ---------------------------------------------------------------------------
// w f32 [K][N]  ->  wT bf16 [N][K]   (LDS-tiled for coalescing)
// ---------------------------------------------------------------------------
__global__ __launch_bounds__(256)
void transpose_bf16_kernel(const float* __restrict__ w,
                           __bf16* __restrict__ wt, int K, int N) {
  __shared__ float tile[32][33];
  int n0 = blockIdx.x * 32, k0 = blockIdx.y * 32;
  int tx = threadIdx.x & 31, ty = threadIdx.x >> 5;
  #pragma unroll
  for (int r = ty; r < 32; r += 8)
    tile[r][tx] = w[(size_t)(k0 + r) * N + n0 + tx];
  __syncthreads();
  #pragma unroll
  for (int r = ty; r < 32; r += 8)
    wt[(size_t)(n0 + r) * K + k0 + tx] = f2bf(tile[tx][r]);
}

// ---------------------------------------------------------------------------
// Tiled WMMA GEMM: C[M][N] = A[M][K] * Bt[N][K]^T, bf16 in, f32 accum.
// 128x128 tile, BK=64, 256 threads = 8 waves, each wave 32x64.
// 2-deep pipeline over double-buffered LDS (72 KB): global fetch for step
// t+2 issues during step t, LDS commit of step t+1 happens one full compute
// iteration after its loads were issued -> the s_wait_loadcnt is fully
// hidden and there is exactly one __syncthreads per K-step.
// MODE 0: C bf16 row-major.  MODE 1: C f32 row-major (final output).
// MODE 2: V-projection, write transposed vt[b][kvh][hd][s] (bf16).
// All problem dims here are multiples of 128 -> no bounds checks.
// ---------------------------------------------------------------------------
template <int MODE>
__global__ __launch_bounds__(256)
void gemm_bf16_kernel(const __bf16* __restrict__ A,
                      const __bf16* __restrict__ Bt,
                      void* __restrict__ Cout, int M, int N, int K) {
  constexpr int LD = 72;                       // 64 + 8 pad (bank spread)
  __shared__ __align__(16) __bf16 sA[2][128 * LD];
  __shared__ __align__(16) __bf16 sB[2][128 * LD];
  int lane = threadIdx.x & 31;
  int wave = threadIdx.x >> 5;
  int wm = wave >> 1, wn = wave & 1;
  int m0 = blockIdx.y * 128, n0 = blockIdx.x * 128;
  int nlo = lane & 15;

  v8f acc[2][4] = {};
  v8bf ra[4], rb[4];

  auto gload = [&](int kt) {
    #pragma unroll
    for (int it = 0; it < 4; ++it) {
      int linear = threadIdx.x + it * 256;
      int r = linear >> 3, c = (linear & 7) * 8;
      ra[it] = *(const v8bf*)(A  + (size_t)(m0 + r) * K + kt + c);
      rb[it] = *(const v8bf*)(Bt + (size_t)(n0 + r) * K + kt + c);
    }
  };
  auto lcommit = [&](int buf) {
    #pragma unroll
    for (int it = 0; it < 4; ++it) {
      int linear = threadIdx.x + it * 256;
      int r = linear >> 3, c = (linear & 7) * 8;
      *(v8bf*)(sA[buf] + r * LD + c) = ra[it];
      *(v8bf*)(sB[buf] + r * LD + c) = rb[it];
    }
  };

  int nk = K >> 6;
  gload(0);
  lcommit(0);
  __syncthreads();
  if (nk > 1) gload(64);

  for (int t = 0; t < nk; ++t) {
    int cur = t & 1;
    if (t + 1 < nk) lcommit(cur ^ 1);          // loads issued 1 iter ago
    if (t + 2 < nk) gload((t + 2) << 6);       // hide latency under WMMAs
    #pragma unroll
    for (int kc = 0; kc < 2; ++kc) {
      v16bf af[2], bfr[4];
      #pragma unroll
      for (int mf = 0; mf < 2; ++mf)
        af[mf] = frag_a(sA[cur] + (wm * 32 + mf * 16 + nlo) * LD + kc * 32, lane);
      #pragma unroll
      for (int nf = 0; nf < 4; ++nf)
        bfr[nf] = frag_b(sB[cur] + (wn * 64 + nf * 16 + nlo) * LD + kc * 32, lane);
      #pragma unroll
      for (int mf = 0; mf < 2; ++mf)
        #pragma unroll
        for (int nf = 0; nf < 4; ++nf)
          acc[mf][nf] = wmma_bf16(af[mf], bfr[nf], acc[mf][nf]);
    }
    __syncthreads();
  }

  // C layout: VGPR slot i -> row i (lanes 0-15) / i+8 (lanes 16-31), col=lane%16
  int half8 = (lane >= 16) ? 8 : 0;
  #pragma unroll
  for (int mf = 0; mf < 2; ++mf)
    #pragma unroll
    for (int nf = 0; nf < 4; ++nf)
      #pragma unroll
      for (int i = 0; i < 8; ++i) {
        int gm = m0 + wm * 32 + mf * 16 + i + half8;
        int gn = n0 + wn * 64 + nf * 16 + nlo;
        float val = acc[mf][nf][i];
        if constexpr (MODE == 0) {
          ((__bf16*)Cout)[(size_t)gm * N + gn] = f2bf(val);
        } else if constexpr (MODE == 1) {
          ((float*)Cout)[(size_t)gm * N + gn] = val;
        } else {
          int bb = gm >> 10, ss = gm & 1023, kvh = gn >> 7, hd = gn & 127;
          ((__bf16*)Cout)[(((size_t)bb * HKV + kvh) * HD + hd) * SEQ + ss] = f2bf(val);
        }
      }
}

// ---------------------------------------------------------------------------
// RoPE over t[b][s][heads][128], interleaved (even,odd) pairs, bf16 in/out.
// ---------------------------------------------------------------------------
__global__ void rope_kernel(__bf16* __restrict__ t,
                            const float* __restrict__ cs,
                            const float* __restrict__ sn,
                            int heads, int total) {
  int idx = blockIdx.x * 256 + threadIdx.x;
  if (idx >= total) return;
  int i = idx & 63;
  int rest = idx >> 6;                 // (b*1024+s)*heads + h
  int h = rest % heads;
  int bs = rest / heads;
  int s = bs & 1023;
  float c = cs[s * 64 + i], si = sn[s * 64 + i];
  __bf16* p = t + ((size_t)bs * heads + h) * 128 + 2 * i;
  float tr = bf2f(p[0]), ti = bf2f(p[1]);
  p[0] = f2bf(tr * c - ti * si);
  p[1] = f2bf(tr * si + ti * c);
}

// ---------------------------------------------------------------------------
// Flash attention (no causal mask). Grid = B*HQ*(SEQ/128), 256 thr = 8 waves.
// All 8 waves share (b, kv-head): each 32-key K tile (32x128) and V tile
// (128x32) is staged ONCE into double-buffered LDS via a 2-deep
// global->reg->LDS pipeline (fetch blk+2 during blk, commit blk+1 at top of
// its own iteration), then every wave feeds its WMMAs from ds_load.
//   scores: 8x WMMA (Q 16x128 * K^T 128x32), K frags preloaded as a clause
//   PV    : 8x WMMA (P 16x32  * V  32x128), V frags preloaded; P staged via
//           per-wave LDS tile to convert C-layout f32 -> A-layout bf16.
// Softmax row stats: DPP16 row_xmask butterflies (no LDS traffic).
// k cache: [b][s][kvh][hd] (row stride 1024); vt: [b][kvh][hd][s].
// ---------------------------------------------------------------------------
__global__ __launch_bounds__(256)
void attn_kernel(const __bf16* __restrict__ q,
                 const __bf16* __restrict__ kc,
                 const __bf16* __restrict__ vt,
                 __bf16* __restrict__ outb) {
  constexpr int LDP = 40;                      // 32 + 8 pad
  constexpr int LDK = 136;                     // 128 + 8 pad (272B rows)
  constexpr int LDV = 40;                      // 32 + 8 pad  (80B rows)
  constexpr int NB  = SEQ / 32;                // 32 key blocks
  __shared__ __align__(16) __bf16 sP[8 * 16 * LDP];
  __shared__ __align__(16) __bf16 sK[2][32 * LDK];
  __shared__ __align__(16) __bf16 sV[2][128 * LDV];

  int lane = threadIdx.x & 31;
  int wave = threadIdx.x >> 5;
  int qt = blockIdx.x & 7;                     // SEQ/128 = 8
  int h  = (blockIdx.x >> 3) & 31;
  int b  = blockIdx.x >> 8;
  int kvh = h >> 2;
  int q0 = qt * 128 + wave * 16;
  int nlo = lane & 15;
  int half8 = (lane >= 16) ? 8 : 0;
  const float scale = 0.08838834764831845f;    // 1/sqrt(128)

  const __bf16* kbase = kc + ((size_t)b * SEQ) * (HKV * HD) + kvh * HD;
  const __bf16* vbase = vt + (((size_t)b * HKV + kvh) * HD) * SEQ;

  v8bf rk[2], rv[2];
  auto gfetch = [&](int blk) {                 // global -> regs (no waits)
    int kb = blk * 32;
    #pragma unroll
    for (int it = 0; it < 2; ++it) {
      int lin = threadIdx.x + it * 256;
      int kr = lin >> 4, kcol = (lin & 15) * 8;     // K: 16 chunks/row
      rk[it] = *(const v8bf*)(kbase + (size_t)(kb + kr) * (HKV * HD) + kcol);
      int vr = lin >> 2, vcol = (lin & 3) * 8;      // V: 4 chunks/row
      rv[it] = *(const v8bf*)(vbase + (size_t)vr * SEQ + kb + vcol);
    }
  };
  auto lcommit = [&](int buf) {                // regs -> LDS
    #pragma unroll
    for (int it = 0; it < 2; ++it) {
      int lin = threadIdx.x + it * 256;
      int kr = lin >> 4, kcol = (lin & 15) * 8;
      *(v8bf*)(sK[buf] + kr * LDK + kcol) = rk[it];
      int vr = lin >> 2, vcol = (lin & 3) * 8;
      *(v8bf*)(sV[buf] + vr * LDV + vcol) = rv[it];
    }
  };

  // Q fragments: 16 rows x 128 k, 4 chunks of 32 (loaded once from global)
  const __bf16* qrow = q + ((size_t)(b * SEQ + q0 + nlo)) * DIM + h * HD;
  v16bf qf[4];
  #pragma unroll
  for (int c = 0; c < 4; ++c) qf[c] = frag_a(qrow + c * 32, lane);

  float mOld[8], lsum[8];
  v8f acc[8] = {};
  #pragma unroll
  for (int i = 0; i < 8; ++i) { mOld[i] = -1e30f; lsum[i] = 0.f; }

  __bf16* pw = sP + wave * 16 * LDP;

  gfetch(0);
  lcommit(0);
  __syncthreads();
  gfetch(1);

  for (int blk = 0; blk < NB; ++blk) {
    int cur = blk & 1;
    if (blk + 1 < NB) lcommit(cur ^ 1);        // data fetched 1 iter ago
    if (blk + 2 < NB) gfetch(blk + 2);         // hide under this iter's math

    // ---- scores S = Q * K^T (16 x 32): preload frags, then WMMA chain ----
    const __bf16* kr0 = sK[cur] + nlo * LDK;
    const __bf16* kr1 = sK[cur] + (16 + nlo) * LDK;
    v16bf kf[8];
    #pragma unroll
    for (int c = 0; c < 4; ++c) {
      kf[c]     = frag_b(kr0 + c * 32, lane);
      kf[4 + c] = frag_b(kr1 + c * 32, lane);
    }
    v8f s0 = {}, s1 = {};
    #pragma unroll
    for (int c = 0; c < 4; ++c) {
      s0 = wmma_bf16(qf[c], kf[c], s0);
      s1 = wmma_bf16(qf[c], kf[4 + c], s1);
    }

    // ---- V fragments for this block (clause of ds_loads) ----
    v16bf vf[8];
    #pragma unroll
    for (int nf = 0; nf < 8; ++nf)
      vf[nf] = frag_b(sV[cur] + (nf * 16 + nlo) * LDV, lane);

    // ---- online softmax: DPP row_xmask reductions over 16-lane halves ----
    float p0[8], p1[8], alpha[8];
    #pragma unroll
    for (int i = 0; i < 8; ++i) {
      float a = s0[i] * scale, c2 = s1[i] * scale;
      float mN = fmaxf(mOld[i], rmax16(fmaxf(a, c2)));
      alpha[i] = __expf(mOld[i] - mN);
      p0[i] = __expf(a - mN);
      p1[i] = __expf(c2 - mN);
      lsum[i] = lsum[i] * alpha[i] + rsum16(p0[i] + p1[i]);
      mOld[i] = mN;
    }
    #pragma unroll
    for (int nf = 0; nf < 8; ++nf)
      #pragma unroll
      for (int i = 0; i < 8; ++i) acc[nf][i] *= alpha[i];

    // ---- P: C-layout f32 -> LDS -> A-layout bf16 (same-wave DS in-order) --
    #pragma unroll
    for (int i = 0; i < 8; ++i) {
      pw[(i + half8) * LDP + nlo]      = f2bf(p0[i]);
      pw[(i + half8) * LDP + 16 + nlo] = f2bf(p1[i]);
    }
    v16bf pf = frag_a(pw + nlo * LDP, lane);

    // ---- acc += P * V ----
    #pragma unroll
    for (int nf = 0; nf < 8; ++nf)
      acc[nf] = wmma_bf16(pf, vf[nf], acc[nf]);

    __syncthreads();   // cur readers done; commits visible for next iter
  }

  // ---- normalize and store attn[b][s][h][hd] ----
  #pragma unroll
  for (int i = 0; i < 8; ++i) {
    float inv = 1.0f / lsum[i];
    int s = q0 + i + half8;
    __bf16* orow = outb + ((size_t)(b * SEQ + s)) * DIM + h * HD;
    #pragma unroll
    for (int nf = 0; nf < 8; ++nf)
      orow[nf * 16 + nlo] = f2bf(acc[nf][i] * inv);
  }
}

// ---------------------------------------------------------------------------
// Launch. Workspace layout (bytes, 256-aligned; ~168 MB total):
//   xb   33.5M (x bf16; reused later as attention output)
//   wqT  33.5M | wkT 8.4M | wvT 8.4M | woT 33.5M
//   qb   33.5M | kb  8.4M | vtb 8.4M
// ---------------------------------------------------------------------------
extern "C" void kernel_launch(void* const* d_in, const int* in_sizes, int n_in,
                              void* d_out, int out_size, void* d_ws, size_t ws_size,
                              hipStream_t stream) {
  const float* x  = (const float*)d_in[0];
  const float* fc = (const float*)d_in[1];
  const float* fs = (const float*)d_in[2];
  const float* wq = (const float*)d_in[3];
  const float* wk = (const float*)d_in[4];
  const float* wv = (const float*)d_in[5];
  const float* wo = (const float*)d_in[6];
  float* out = (float*)d_out;

  char* ws = (char*)d_ws;
  size_t off = 0;
  auto carve = [&](size_t bytes) {
    void* p = ws + off;
    off += (bytes + 255) & ~(size_t)255;
    return p;
  };
  const size_t ACT = (size_t)BATCH * SEQ * DIM * 2;   // 33.5 MB
  const size_t KV  = (size_t)BATCH * SEQ * 1024 * 2;  // 8.4 MB
  __bf16* xb  = (__bf16*)carve(ACT);                  // x bf16, reused as attn out
  __bf16* wqT = (__bf16*)carve((size_t)DIM * DIM * 2);
  __bf16* wkT = (__bf16*)carve((size_t)1024 * DIM * 2);
  __bf16* wvT = (__bf16*)carve((size_t)1024 * DIM * 2);
  __bf16* woT = (__bf16*)carve((size_t)DIM * DIM * 2);
  __bf16* qb  = (__bf16*)carve(ACT);
  __bf16* kb  = (__bf16*)carve(KV);
  __bf16* vtb = (__bf16*)carve(KV);
  (void)ws_size; (void)in_sizes; (void)n_in; (void)out_size;

  const int M = BATCH * SEQ;                          // 4096 rows

  // 1) precision conversion / weight transposes (bandwidth-trivial)
  int n4 = M * DIM / 4;
  cast_bf16_kernel<<<(n4 + 255) / 256, 256, 0, stream>>>(x, xb, n4);
  transpose_bf16_kernel<<<dim3(DIM / 32,  DIM / 32), 256, 0, stream>>>(wq, wqT, DIM, DIM);
  transpose_bf16_kernel<<<dim3(1024 / 32, DIM / 32), 256, 0, stream>>>(wk, wkT, DIM, 1024);
  transpose_bf16_kernel<<<dim3(1024 / 32, DIM / 32), 256, 0, stream>>>(wv, wvT, DIM, 1024);
  transpose_bf16_kernel<<<dim3(DIM / 32,  DIM / 32), 256, 0, stream>>>(wo, woT, DIM, DIM);

  // 2) QKV projections (V written pre-transposed for the PV WMMA B-operand)
  gemm_bf16_kernel<0><<<dim3(DIM / 128,  M / 128), 256, 0, stream>>>(xb, wqT, qb,  M, DIM,  DIM);
  gemm_bf16_kernel<0><<<dim3(1024 / 128, M / 128), 256, 0, stream>>>(xb, wkT, kb,  M, 1024, DIM);
  gemm_bf16_kernel<2><<<dim3(1024 / 128, M / 128), 256, 0, stream>>>(xb, wvT, vtb, M, 1024, DIM);

  // 3) RoPE on Q and K
  rope_kernel<<<(M * HQ  * 64 + 255) / 256, 256, 0, stream>>>(qb, fc, fs, HQ,  M * HQ  * 64);
  rope_kernel<<<(M * HKV * 64 + 255) / 256, 256, 0, stream>>>(kb, fc, fs, HKV, M * HKV * 64);

  // 4) flash attention -> reuse xb as attn output buffer
  attn_kernel<<<BATCH * HQ * (SEQ / 128), 256, 0, stream>>>(qb, kb, vtb, xb);

  // 5) output projection, f32 straight to d_out
  gemm_bf16_kernel<1><<<dim3(DIM / 128, M / 128), 256, 0, stream>>>(xb, woT, out, M, DIM, DIM);
}
